// NSTransformer_9208409882818
// MI455X (gfx1250) — matrix operand order repaired and checked by reading, not compiled, not run on hardware
//
#include <hip/hip_runtime.h>
#include <math.h>

// ---------------- model dimensions (from reference) ----------------
constexpr int NB    = 16;      // batch
constexpr int SEQ   = 512;     // L
constexpr int EIN   = 21;      // enc_in
constexpr int DIM   = 512;     // D
constexpr int NH    = 8;       // heads
constexpr int DHD   = 64;      // head dim
constexpr int DFFC  = 2048;    // FFN hidden
constexpr int PREDN = 96;      // pred_len
constexpr int COUT  = 21;      // c_out
constexpr int NTOK  = NB * SEQ;
constexpr float EPSF = 1e-5f;

typedef __attribute__((ext_vector_type(16))) _Float16 v16h;
typedef __attribute__((ext_vector_type(8)))  float    v8f;

// =====================================================================
// WMMA fragment loaders (wave32; A and B use contiguous k = 16*(lane/16) + s)
// =====================================================================

// A fragment 16x32 (MxK), row-major source, tile origin p, leading dim ld.
// lane<16: row=lane, K = {kb..kb+15} with kb=0;
// lane>=16: same row set with kb=16.
__device__ __forceinline__ v16h load_a16x32(const _Float16* __restrict__ p,
                                            int ld, int lane) {
  const int r  = lane & 15;
  const int kb = (lane >> 4) << 4;
  const _Float16* q = p + (size_t)r * ld + kb;
  v16h f;
#pragma unroll
  for (int i = 0; i < 8; ++i) { f[i] = q[i]; f[i + 8] = q[i + 8]; }
  return f;
}

// B fragment 32x16 (KxN) built from W stored as (N x K) row-major, so
// B[k][n] = W[n][k].  lane<16: col n=lane, K=0..15; lane>=16: K=16..31.
__device__ __forceinline__ v16h load_b_wT(const _Float16* __restrict__ w,
                                          int ld, int lane) {
  const int n  = lane & 15;
  const int ks = (lane >> 4) << 4;
  const _Float16* q = w + (size_t)n * ld + ks;
  v16h f;
#pragma unroll
  for (int j = 0; j < 16; ++j) f[j] = q[j];
  return f;
}

// B fragment 32x16 from a (K x N) row-major matrix (e.g. V in attention).
__device__ __forceinline__ v16h load_b_kn(const _Float16* __restrict__ p,
                                          int ld, int lane) {
  const int n  = lane & 15;
  const int ks = (lane >> 4) << 4;
  v16h f;
#pragma unroll
  for (int j = 0; j < 16; ++j) f[j] = p[(size_t)(ks + j) * ld + n];
  return f;
}

__device__ __forceinline__ v8f wmma16(v16h a, v16h b, v8f c) {
  return __builtin_amdgcn_wmma_f32_16x16x32_f16(
      /*neg_a=*/false, a, /*neg_b=*/false, b,
      /*c_mod=*/(short)0, c, /*reuse_a=*/false, /*reuse_b=*/false);
}

// C/D f32 16x16 store: vgpr r holds (m = r + 8*(lane>=16), n = lane&15)
__device__ __forceinline__ void store_tile_f32f16(
    v8f acc, float* __restrict__ C, _Float16* __restrict__ Ch,
    const float* __restrict__ bias, int relu,
    int m0, int n0, int ldc, int lane) {
  const int mh = (lane >> 4) << 3;
  const int n  = n0 + (lane & 15);
#pragma unroll
  for (int r = 0; r < 8; ++r) {
    const int m = m0 + mh + r;
    float v = acc[r] + (bias ? bias[n] : 0.0f);
    if (relu) v = fmaxf(v, 0.0f);
    const size_t idx = (size_t)m * ldc + n;
    if (C)  C[idx]  = v;
    if (Ch) Ch[idx] = (_Float16)v;
  }
}

// =====================================================================
// GEMM: Y[M,N] = X[M,K] * W[N,K]^T (+bias) (+relu); f16 in, f32 acc.
// 4 waves/block, each wave a 32x32 tile -> 64x64 block tile.
// =====================================================================
__global__ __launch_bounds__(128) void wmma_gemm_xwT(
    const _Float16* __restrict__ A, const _Float16* __restrict__ W,
    const float* __restrict__ bias, float* __restrict__ C,
    _Float16* __restrict__ Ch, int M, int N, int K, int relu) {
  const int lane = threadIdx.x & 31;
  const int wave = threadIdx.x >> 5;
  const int m0 = blockIdx.y * 64 + (wave >> 1) * 32;
  const int n0 = blockIdx.x * 64 + (wave & 1) * 32;
  if (m0 >= M || n0 >= N) return;             // wave-uniform

  v8f a00 = {}, a01 = {}, a10 = {}, a11 = {};
  const _Float16* Ar0 = A + (size_t)m0 * K;
  const _Float16* Ar1 = A + (size_t)(m0 + 16) * K;
  const _Float16* Wr0 = W + (size_t)n0 * K;
  const _Float16* Wr1 = W + (size_t)(n0 + 16) * K;

  for (int k0 = 0; k0 < K; k0 += 32) {
    __builtin_prefetch(Ar0 + k0 + 128, 0, 1);   // global_prefetch_b8
    v16h fa0 = load_a16x32(Ar0 + k0, K, lane);
    v16h fa1 = load_a16x32(Ar1 + k0, K, lane);
    v16h fb0 = load_b_wT(Wr0 + k0, K, lane);
    v16h fb1 = load_b_wT(Wr1 + k0, K, lane);
    a00 = wmma16(fa0, fb0, a00);
    a01 = wmma16(fa0, fb1, a01);
    a10 = wmma16(fa1, fb0, a10);
    a11 = wmma16(fa1, fb1, a11);
  }
  store_tile_f32f16(a00, C, Ch, bias, relu, m0,      n0,      N, lane);
  store_tile_f32f16(a01, C, Ch, bias, relu, m0,      n0 + 16, N, lane);
  store_tile_f32f16(a10, C, Ch, bias, relu, m0 + 16, n0,      N, lane);
  store_tile_f32f16(a11, C, Ch, bias, relu, m0 + 16, n0 + 16, N, lane);
}

// =====================================================================
// Attention scores: S[b,h,l,s] = f16( (q.k * tau[b] + delta[b]) / sqrt(dh) )
// =====================================================================
__global__ __launch_bounds__(128) void wmma_attn_scores(
    const _Float16* __restrict__ Qh, const _Float16* __restrict__ Kh,
    _Float16* __restrict__ S, const float* __restrict__ tau,
    const float* __restrict__ delta) {
  const int lane = threadIdx.x & 31;
  const int wave = threadIdx.x >> 5;
  const int bh = blockIdx.z;
  const int b  = bh >> 3;            // NH == 8
  const int h  = bh & 7;
  const _Float16* Qp = Qh + (size_t)b * SEQ * DIM + h * DHD;
  const _Float16* Kp = Kh + (size_t)b * SEQ * DIM + h * DHD;
  _Float16* Sp = S + (size_t)bh * SEQ * SEQ;
  const int m0 = blockIdx.y * 64 + (wave >> 1) * 32;
  const int n0 = blockIdx.x * 64 + (wave & 1) * 32;

  v8f a00 = {}, a01 = {}, a10 = {}, a11 = {};
#pragma unroll
  for (int k0 = 0; k0 < DHD; k0 += 32) {
    v16h fa0 = load_a16x32(Qp + (size_t)m0 * DIM + k0, DIM, lane);
    v16h fa1 = load_a16x32(Qp + (size_t)(m0 + 16) * DIM + k0, DIM, lane);
    v16h fb0 = load_b_wT(Kp + (size_t)n0 * DIM + k0, DIM, lane);
    v16h fb1 = load_b_wT(Kp + (size_t)(n0 + 16) * DIM + k0, DIM, lane);
    a00 = wmma16(fa0, fb0, a00);
    a01 = wmma16(fa0, fb1, a01);
    a10 = wmma16(fa1, fb0, a10);
    a11 = wmma16(fa1, fb1, a11);
  }
  const float t  = tau[b];
  const float dl = delta ? delta[b] : 0.0f;
  const int mh = (lane >> 4) << 3;
  const int nn = lane & 15;
  v8f accs[4] = {a00, a01, a10, a11};
#pragma unroll
  for (int q = 0; q < 4; ++q) {
    const int mb = m0 + ((q >> 1) << 4);
    const int nb = n0 + ((q & 1) << 4) + nn;
#pragma unroll
    for (int r = 0; r < 8; ++r) {
      const float v = (accs[q][r] * t + dl) * 0.125f;  // 1/sqrt(64)
      Sp[(size_t)(mb + mh + r) * SEQ + nb] = (_Float16)v;
    }
  }
}

// =====================================================================
// O[b,l,h*64+d] = sum_s P[b,h,l,s] * V[b,s,h*64+d]   (P f16 in S)
// =====================================================================
__global__ __launch_bounds__(128) void wmma_attn_av(
    const _Float16* __restrict__ S, const _Float16* __restrict__ Vh,
    float* __restrict__ O) {
  const int lane = threadIdx.x & 31;
  const int wave = threadIdx.x >> 5;
  const int bh = blockIdx.z;
  const int b  = bh >> 3;
  const int h  = bh & 7;
  const _Float16* Sp = S + (size_t)bh * SEQ * SEQ;
  const _Float16* Vp = Vh + (size_t)b * SEQ * DIM + h * DHD;
  float* Op = O + (size_t)b * SEQ * DIM + h * DHD;
  const int m0 = blockIdx.y * 64 + (wave >> 1) * 32;
  const int n0 = (wave & 1) * 32;                     // DHD == 64

  v8f a00 = {}, a01 = {}, a10 = {}, a11 = {};
  for (int k0 = 0; k0 < SEQ; k0 += 32) {
    __builtin_prefetch(Sp + (size_t)m0 * SEQ + k0 + 128, 0, 1);
    v16h fa0 = load_a16x32(Sp + (size_t)m0 * SEQ + k0, SEQ, lane);
    v16h fa1 = load_a16x32(Sp + (size_t)(m0 + 16) * SEQ + k0, SEQ, lane);
    v16h fb0 = load_b_kn(Vp + (size_t)k0 * DIM + n0, DIM, lane);
    v16h fb1 = load_b_kn(Vp + (size_t)k0 * DIM + n0 + 16, DIM, lane);
    a00 = wmma16(fa0, fb0, a00);
    a01 = wmma16(fa0, fb1, a01);
    a10 = wmma16(fa1, fb0, a10);
    a11 = wmma16(fa1, fb1, a11);
  }
  const int mh = (lane >> 4) << 3;
  const int nn = lane & 15;
  v8f accs[4] = {a00, a01, a10, a11};
#pragma unroll
  for (int q = 0; q < 4; ++q) {
    const int mb = m0 + ((q >> 1) << 4);
    const int nb = n0 + ((q & 1) << 4) + nn;
#pragma unroll
    for (int r = 0; r < 8; ++r)
      Op[(size_t)(mb + mh + r) * DIM + nb] = accs[q][r];
  }
}

// =====================================================================
// elementwise / reduction glue kernels
// =====================================================================
__global__ __launch_bounds__(256) void convert_f32_f16(
    const float* __restrict__ s, _Float16* __restrict__ d, size_t n) {
  size_t i = (size_t)blockIdx.x * blockDim.x + threadIdx.x;
  if (i < n) d[i] = (_Float16)s[i];
}

// per (b,e): mean / std over L
__global__ __launch_bounds__(64) void stats_kernel(
    const float* __restrict__ x, float* __restrict__ meanb,
    float* __restrict__ stdb) {
  __shared__ float red[64];
  const int be = blockIdx.x, b = be / EIN, e = be % EIN;
  float s = 0.f, ss = 0.f;
  for (int l = threadIdx.x; l < SEQ; l += 64) {
    float v = x[((size_t)b * SEQ + l) * EIN + e];
    s += v; ss += v * v;
  }
  red[threadIdx.x] = s; __syncthreads();
  for (int k = 32; k > 0; k >>= 1) { if (threadIdx.x < k) red[threadIdx.x] += red[threadIdx.x + k]; __syncthreads(); }
  const float sum = red[0]; __syncthreads();
  red[threadIdx.x] = ss; __syncthreads();
  for (int k = 32; k > 0; k >>= 1) { if (threadIdx.x < k) red[threadIdx.x] += red[threadIdx.x + k]; __syncthreads(); }
  if (threadIdx.x == 0) {
    const float mu = sum / SEQ;
    const float var = red[0] / SEQ - mu * mu;
    meanb[be] = mu;
    stdb[be]  = sqrtf(var + EPSF);
  }
}

__global__ __launch_bounds__(256) void normalize_kernel(
    const float* __restrict__ x, const float* __restrict__ meanb,
    const float* __restrict__ stdb, float* __restrict__ xn) {
  size_t i = (size_t)blockIdx.x * blockDim.x + threadIdx.x;
  if (i >= (size_t)NB * SEQ * EIN) return;
  const int e = (int)(i % EIN);
  const int b = (int)(i / ((size_t)SEQ * EIN));
  xn[i] = (x[i] - meanb[b * EIN + e]) / stdb[b * EIN + e];
}

// projector (tau / delta): conv over E_IN axis + 42->256->256->1 MLP
__global__ __launch_bounds__(256) void projector_kernel(
    const float* __restrict__ xr, const float* __restrict__ stats,
    const float* __restrict__ cw, const float* __restrict__ w0,
    const float* __restrict__ b0, const float* __restrict__ w1,
    const float* __restrict__ b1, const float* __restrict__ ow,
    float* __restrict__ out, int do_exp) {
  __shared__ float sin_[2 * EIN];
  __shared__ float sh0[256];
  __shared__ float sh1[256];
  __shared__ float red[256];
  const int b = blockIdx.x, t = threadIdx.x;
  if (t < EIN) {
    float s = 0.f;
    for (int l = 0; l < SEQ; ++l) {
      const float* xp = xr + ((size_t)b * SEQ + l) * EIN;
      const float* wp = cw + l * 3;
#pragma unroll
      for (int k = 0; k < 3; ++k)
        s += xp[(t + k - 1 + EIN) % EIN] * wp[k];
    }
    sin_[t] = s;
  }
  if (t >= EIN && t < 2 * EIN) sin_[t] = stats[b * EIN + (t - EIN)];
  __syncthreads();
  {
    float s = b0[t];
    for (int i = 0; i < 2 * EIN; ++i) s += sin_[i] * w0[t * 2 * EIN + i];
    sh0[t] = fmaxf(s, 0.f);
  }
  __syncthreads();
  {
    float s = b1[t];
    for (int i = 0; i < 256; ++i) s += sh0[i] * w1[t * 256 + i];
    sh1[t] = fmaxf(s, 0.f);
  }
  __syncthreads();
  red[t] = sh1[t] * ow[t]; __syncthreads();
  for (int k = 128; k > 0; k >>= 1) { if (t < k) red[t] += red[t + k]; __syncthreads(); }
  if (t == 0) out[b] = do_exp ? expf(red[0]) : red[0];
}

// token conv embedding (circular over L) + sinusoidal PE
__global__ __launch_bounds__(256) void embed_kernel(
    const float* __restrict__ xn, const float* __restrict__ ew,
    float* __restrict__ X) {
  size_t i = (size_t)blockIdx.x * blockDim.x + threadIdx.x;
  if (i >= (size_t)NTOK * DIM) return;
  const int d  = (int)(i % DIM);
  const int bl = (int)(i / DIM);
  const int l  = bl % SEQ;
  const int b  = bl / SEQ;
  float s = 0.f;
  for (int c = 0; c < EIN; ++c) {
    const float* wp = ew + ((size_t)d * EIN + c) * 3;
#pragma unroll
    for (int k = 0; k < 3; ++k) {
      const int lm = (l + k - 1 + SEQ) % SEQ;
      s += xn[((size_t)b * SEQ + lm) * EIN + c] * wp[k];
    }
  }
  const int   i2  = d & ~1;
  const float div = expf(-logf(10000.0f) * (float)i2 / (float)DIM);
  const float ang = (float)l * div;
  s += (d & 1) ? cosf(ang) : sinf(ang);
  X[i] = s;
}

// in-place row softmax over SEQ f16 values
__global__ __launch_bounds__(256) void softmax_rows(_Float16* __restrict__ S) {
  __shared__ float red[256];
  _Float16* row = S + (size_t)blockIdx.x * SEQ;
  const int t = threadIdx.x;
  float v0 = (float)row[t], v1 = (float)row[t + 256];
  red[t] = fmaxf(v0, v1); __syncthreads();
  for (int k = 128; k > 0; k >>= 1) { if (t < k) red[t] = fmaxf(red[t], red[t + k]); __syncthreads(); }
  const float m = red[0]; __syncthreads();
  v0 = __expf(v0 - m); v1 = __expf(v1 - m);
  red[t] = v0 + v1; __syncthreads();
  for (int k = 128; k > 0; k >>= 1) { if (t < k) red[t] += red[t + k]; __syncthreads(); }
  const float inv = 1.0f / red[0];
  row[t]       = (_Float16)(v0 * inv);
  row[t + 256] = (_Float16)(v1 * inv);
}

// X = LayerNorm(X + Tadd) * g + b   (Tadd nullable), row = token, dim = DIM
__global__ __launch_bounds__(256) void ln_fused_kernel(
    float* __restrict__ X, const float* __restrict__ Tadd,
    const float* __restrict__ g, const float* __restrict__ bt) {
  __shared__ float red[256];
  const size_t base = (size_t)blockIdx.x * DIM;
  const int t = threadIdx.x;
  float v0 = X[base + t]       + (Tadd ? Tadd[base + t]       : 0.f);
  float v1 = X[base + t + 256] + (Tadd ? Tadd[base + t + 256] : 0.f);
  red[t] = v0 + v1; __syncthreads();
  for (int k = 128; k > 0; k >>= 1) { if (t < k) red[t] += red[t + k]; __syncthreads(); }
  const float mu = red[0] / DIM; __syncthreads();
  const float d0 = v0 - mu, d1 = v1 - mu;
  red[t] = d0 * d0 + d1 * d1; __syncthreads();
  for (int k = 128; k > 0; k >>= 1) { if (t < k) red[t] += red[t + k]; __syncthreads(); }
  const float r = rsqrtf(red[0] / DIM + EPSF);
  X[base + t]       = d0 * r * g[t]       + bt[t];
  X[base + t + 256] = d1 * r * g[t + 256] + bt[t + 256];
}

// dec_proj: DEC[b,l,c] = sum_d X[b,l,d]*W[c,d] + bias[c]
__global__ __launch_bounds__(256) void dec_proj_kernel(
    const float* __restrict__ X, const float* __restrict__ W,
    const float* __restrict__ bias, float* __restrict__ DEC) {
  size_t i = (size_t)blockIdx.x * blockDim.x + threadIdx.x;
  if (i >= (size_t)NTOK * COUT) return;
  const int c   = (int)(i % COUT);
  const size_t row = i / COUT;
  float s = bias[c];
  const float* xp = X + row * DIM;
  const float* wp = W + (size_t)c * DIM;
  for (int d = 0; d < DIM; ++d) s += xp[d] * wp[d];
  DEC[i] = s;
}

__global__ __launch_bounds__(256) void descale_kernel(
    float* __restrict__ DEC, const float* __restrict__ stdb,
    const float* __restrict__ meanb) {
  size_t i = (size_t)blockIdx.x * blockDim.x + threadIdx.x;
  if (i >= (size_t)NTOK * COUT) return;
  const int c = (int)(i % COUT);
  const int b = (int)(i / ((size_t)SEQ * COUT));
  DEC[i] = DEC[i] * stdb[b * EIN + c] + meanb[b * EIN + c];
}

// out[b,p,c] = sum_l DEC[b,l,c] * proj1_w[p,l] + proj1_b[p]
__global__ __launch_bounds__(256) void finalproj_kernel(
    const float* __restrict__ DEC, const float* __restrict__ PW,
    const float* __restrict__ PB, float* __restrict__ out) {
  size_t i = (size_t)blockIdx.x * blockDim.x + threadIdx.x;
  if (i >= (size_t)NB * PREDN * COUT) return;
  const int c = (int)(i % COUT);
  const int p = (int)((i / COUT) % PREDN);
  const int b = (int)(i / ((size_t)PREDN * COUT));
  float s = PB[p];
  const float* dp = DEC + (size_t)b * SEQ * COUT + c;
  const float* wp = PW + (size_t)p * SEQ;
  for (int l = 0; l < SEQ; ++l) s += dp[(size_t)l * COUT] * wp[l];
  out[i] = s;
}

// =====================================================================
// host orchestration
// =====================================================================
extern "C" void kernel_launch(void* const* d_in, const int* in_sizes, int n_in,
                              void* d_out, int out_size, void* d_ws, size_t ws_size,
                              hipStream_t stream) {
  // d_in layout assumption: insertion-order flattening of setup_inputs():
  // 0:x_enc 1:emb_w 2..7:tau{conv_w,w0,b0,w1,b1,out_w} 8..13:delta{...}
  // 14..29:enc_layer0{wq,bq,wk,bk,wv,bv,wo,bo,w1,b1,w2,b2,ln1g,ln1b,ln2g,ln2b}
  // 30..45:enc_layer1  46,47:enc_norm{g,b}
  // 48..73:dec_layer{self(8),cross(8),w1,b1,w2,b2,ln1g,ln1b,ln2g,ln2b,ln3g,ln3b}
  // 74,75:dec_norm  76,77:dec_proj{w,b}  78,79:proj1{w,b}
  auto F = [&](int i) { return (const float*)d_in[i]; };
  const float* x_enc = F(0);

  // ---- workspace bump allocator (needs ~190 MB) ----
  size_t off = 0;
  auto alloc = [&](size_t bytes) -> void* {
    off = (off + 255) & ~(size_t)255;
    void* p = (char*)d_ws + off;
    off += bytes;
    return p;
  };
  float* meanb  = (float*)alloc(NB * EIN * 4);
  float* stdb   = (float*)alloc(NB * EIN * 4);
  float* tauv   = (float*)alloc(NB * 4);
  float* deltav = (float*)alloc(NB * 4);
  float* xnorm  = (float*)alloc((size_t)NB * SEQ * EIN * 4);
  float* X      = (float*)alloc((size_t)NTOK * DIM * 4);
  float* T      = (float*)alloc((size_t)NTOK * DIM * 4);
  float* ENC    = (float*)alloc((size_t)NTOK * DIM * 4);
  _Float16* Xh  = (_Float16*)alloc((size_t)NTOK * DIM * 2);
  _Float16* Qh  = (_Float16*)alloc((size_t)NTOK * DIM * 2);
  _Float16* Kh  = (_Float16*)alloc((size_t)NTOK * DIM * 2);
  _Float16* Vh  = (_Float16*)alloc((size_t)NTOK * DIM * 2);
  _Float16* Sc  = (_Float16*)alloc((size_t)NB * NH * SEQ * SEQ * 2);
  _Float16* Hff = (_Float16*)alloc((size_t)NTOK * DFFC * 2);
  _Float16* Wh  = (_Float16*)alloc((size_t)DIM * DFFC * 2);
  float* DEC    = (float*)alloc((size_t)NTOK * COUT * 4);
  (void)ws_size; (void)n_in; (void)in_sizes; (void)out_size;

  auto conv = [&](const float* s, _Float16* d, size_t n) {
    convert_f32_f16<<<dim3((unsigned)((n + 255) / 256)), 256, 0, stream>>>(s, d, n);
  };
  // Y = A(f16, MxK) * W(f32->f16, NxK)^T + bias
  auto gemm = [&](const _Float16* A, const float* Wsrc, size_t wn,
                  const float* bias, float* C, _Float16* Ch,
                  int M, int N, int K, int relu) {
    conv(Wsrc, Wh, wn);
    dim3 g(N / 64, M / 64, 1);
    wmma_gemm_xwT<<<g, 128, 0, stream>>>(A, Wh, bias, C, Ch, M, N, K, relu);
  };
  // T = DS-Attention(xq, xkv); weights at wb..wb+7 = wq,bq,wk,bk,wv,bv,wo,bo
  auto attention = [&](const float* xq, const float* xkv, int wb,
                       const float* taup, const float* deltap) {
    conv(xq, Xh, (size_t)NTOK * DIM);
    gemm(Xh, F(wb + 0), (size_t)DIM * DIM, F(wb + 1), nullptr, Qh, NTOK, DIM, DIM, 0);
    if (xkv != xq) conv(xkv, Xh, (size_t)NTOK * DIM);
    gemm(Xh, F(wb + 2), (size_t)DIM * DIM, F(wb + 3), nullptr, Kh, NTOK, DIM, DIM, 0);
    gemm(Xh, F(wb + 4), (size_t)DIM * DIM, F(wb + 5), nullptr, Vh, NTOK, DIM, DIM, 0);
    dim3 gs(SEQ / 64, SEQ / 64, NB * NH);
    wmma_attn_scores<<<gs, 128, 0, stream>>>(Qh, Kh, Sc, taup, deltap);
    softmax_rows<<<NB * NH * SEQ, 256, 0, stream>>>(Sc);
    dim3 ga(1, SEQ / 64, NB * NH);
    wmma_attn_av<<<ga, 128, 0, stream>>>(Sc, Vh, T);
    conv(T, Xh, (size_t)NTOK * DIM);
    gemm(Xh, F(wb + 6), (size_t)DIM * DIM, F(wb + 7), T, nullptr, NTOK, DIM, DIM, 0);
  };
  auto ffn = [&](int iw1) {  // T = FFN(X); weights iw1..iw1+3 = w1,b1,w2,b2
    conv(X, Xh, (size_t)NTOK * DIM);
    gemm(Xh, F(iw1 + 0), (size_t)DFFC * DIM, F(iw1 + 1), nullptr, Hff, NTOK, DFFC, DIM, 1);
    gemm(Hff, F(iw1 + 2), (size_t)DIM * DFFC, F(iw1 + 3), T, nullptr, NTOK, DIM, DFFC, 0);
  };
  auto ln = [&](const float* add, int ig, int ib) {
    ln_fused_kernel<<<NTOK, 256, 0, stream>>>(X, add, F(ig), F(ib));
  };

  // ---- de-stationarization ----
  stats_kernel<<<NB * EIN, 64, 0, stream>>>(x_enc, meanb, stdb);
  normalize_kernel<<<(NB * SEQ * EIN + 255) / 256, 256, 0, stream>>>(x_enc, meanb, stdb, xnorm);
  projector_kernel<<<NB, 256, 0, stream>>>(x_enc, stdb,  F(2), F(3), F(4), F(5),  F(6),  F(7),  tauv,   1);
  projector_kernel<<<NB, 256, 0, stream>>>(x_enc, meanb, F(8), F(9), F(10), F(11), F(12), F(13), deltav, 0);

  // ---- embedding ----
  embed_kernel<<<(int)(((size_t)NTOK * DIM + 255) / 256), 256, 0, stream>>>(xnorm, F(1), X);

  // ---- encoder (2 layers) ----
  for (int li = 0; li < 2; ++li) {
    const int Lb = 14 + li * 16;
    attention(X, X, Lb, tauv, deltav);
    ln(T, Lb + 12, Lb + 13);
    ffn(Lb + 8);
    ln(T, Lb + 14, Lb + 15);
  }
  ln(nullptr, 46, 47);
  hipMemcpyAsync(ENC, X, (size_t)NTOK * DIM * 4, hipMemcpyDeviceToDevice, stream);

  // ---- decoder (1 layer) ----
  attention(X, X, 48, tauv, nullptr);      // self-attn: delta = 0
  ln(T, 68, 69);
  attention(X, ENC, 56, tauv, deltav);     // cross-attn
  ln(T, 70, 71);
  ffn(64);
  ln(T, 72, 73);
  ln(nullptr, 74, 75);

  // ---- head ----
  dec_proj_kernel<<<(int)(((size_t)NTOK * COUT + 255) / 256), 256, 0, stream>>>(X, F(76), F(77), DEC);
  descale_kernel<<<(int)(((size_t)NTOK * COUT + 255) / 256), 256, 0, stream>>>(DEC, stdb, meanb);
  finalproj_kernel<<<(NB * PREDN * COUT + 255) / 256, 256, 0, stream>>>(DEC, F(78), F(79), (float*)d_out);
}
